// PhCNN_26594437496937
// MI455X (gfx1250) — matrix operand order repaired and checked by reading, not compiled, not run on hardware
//
#include <hip/hip_runtime.h>
#include <math.h>

// ---------------------------------------------------------------------------
// Types for CDNA5 WMMA (wave32): V_WMMA_F32_16X16X32_BF16
// ---------------------------------------------------------------------------
typedef __attribute__((ext_vector_type(16))) __bf16 v16bf;
typedef __attribute__((ext_vector_type(8)))  __bf16 v8bf;
typedef __attribute__((ext_vector_type(8)))  float  v8f;

// Problem constants (from reference)
#define F_FEAT   2048
#define D_COORD  306
#define D_PAD    320      // 306 padded to multiple of 32 for K-loop
#define BATCH    128
#define NFILT    16
#define HIDDEN   64
#define NCLS     10
#define DENSE_IN 16384    // 16 * 1024
#define K_STEPS  10       // 320 / 32
#define N_TILES  (F_FEAT / 16)   // 128 g-tiles

__device__ __forceinline__ unsigned short f2bfbits(float f) {
    unsigned u = __float_as_uint(f);
    unsigned r = u + 0x7FFFu + ((u >> 16) & 1u);   // round-to-nearest-even
    return (unsigned short)(r >> 16);
}
__device__ __forceinline__ __bf16 bfFromBits(unsigned short b) {
    return __builtin_bit_cast(__bf16, b);
}
__device__ __forceinline__ float bfBitsToF(unsigned short b) {
    return __uint_as_float(((unsigned)b) << 16);
}
__device__ __forceinline__ __bf16 f2bf(float f) { return bfFromBits(f2bfbits(f)); }

// Load one 32-wide K-step of the A operand (16x32 bf16, documented layout):
// lane holds row (base row + l15); k bytes at ko + half*8 + {0..7} and +16..23
__device__ __forceinline__ v16bf load_a_step(const __bf16* aRow, int ko) {
    v8bf lo = *(const v8bf*)(aRow + ko);
    v8bf hi = *(const v8bf*)(aRow + ko + 16);
    v16bf v;
#pragma unroll
    for (int i = 0; i < 8; ++i) { v[i] = lo[i]; v[8 + i] = hi[i]; }
    return v;
}
// B operand (32x16 bf16): lane holds column l15 (a row of the row-major src);
// k at ko + half*8 + {0..7} and +16..23 -> two 16B loads (matches A)
__device__ __forceinline__ v16bf load_b_step(const __bf16* bRow, int ko) {
    v8bf lo = *(const v8bf*)(bRow + ko);
    v8bf hi = *(const v8bf*)(bRow + ko + 16);
    v16bf v;
#pragma unroll
    for (int i = 0; i < 8; ++i) { v[i] = lo[i]; v[8 + i] = hi[i]; }
    return v;
}
__device__ __forceinline__ void load_b_tile(const __bf16* bRow, v16bf (&bv)[K_STEPS]) {
#pragma unroll
    for (int s = 0; s < K_STEPS; ++s) bv[s] = load_b_step(bRow, s * 32);
}

// Branchless top-4 insert (ascending scores). Strict '<': a tie keeps the
// incumbent, which is the lower index for in-lane candidates (g monotone).
__device__ __forceinline__ void top4_insert(float (&s)[4], int (&ix)[4],
                                            float v, int g) {
#pragma unroll
    for (int k = 0; k < 4; ++k) {
        const bool c = v < s[k];
        const float sv = s[k];
        const int   si = ix[k];
        s[k]  = c ? v : sv;
        ix[k] = c ? g : si;
        v     = c ? sv : v;
        g     = c ? si : g;
    }
}
// Lexicographic (score, index) variant for the cross-lane merge
__device__ __forceinline__ void top4_insert_lex(float (&s)[4], int (&ix)[4],
                                                float v, int g) {
#pragma unroll
    for (int k = 0; k < 4; ++k) {
        const bool c = (v < s[k]) || (v == s[k] && g < ix[k]);
        const float sv = s[k];
        const int   si = ix[k];
        s[k]  = c ? v : sv;
        ix[k] = c ? g : si;
        v     = c ? sv : v;
        g     = c ? si : g;
    }
}

// ---------------------------------------------------------------------------
// K0: pack coord (D_COORD,1,F_FEAT) -> A0[f][d] bf16 (F_FEAT x D_PAD, zero pad)
//     and sq0[f] = sum_d bf16(coord)^2 (quantized: ordering-consistent)
// ---------------------------------------------------------------------------
__global__ void k0_pack_coord(const float* __restrict__ coord,
                              __bf16* __restrict__ A0, float* __restrict__ sq0) {
    const int f = blockIdx.x;
    const int tid = threadIdx.x;
    float part = 0.f;
    for (int d = tid; d < D_PAD; d += 64) {
        if (d < D_COORD) {
            unsigned short b = f2bfbits(coord[(size_t)d * F_FEAT + f]);
            A0[(size_t)f * D_PAD + d] = bfFromBits(b);
            float q = bfBitsToF(b);
            part += q * q;
        } else {
            A0[(size_t)f * D_PAD + d] = bfFromBits(0);
        }
    }
    __shared__ float red[64];
    red[tid] = part;
    __syncthreads();
    for (int s = 32; s > 0; s >>= 1) {
        if (tid < s) red[tid] += red[tid + s];
        __syncthreads();
    }
    if (tid == 0) sq0[f] = red[0];
}

// ---------------------------------------------------------------------------
// Fused Gram (A·A^T, bf16 WMMA, f32 accumulate) + per-row top-4.
// One wave per 16 f-rows per channel. Ping-pong double-buffered B tiles so
// each load clause is covered by a full tile of WMMA + merge work; per-lane
// register top-4 (lane owns 8 rows x one g-residue class) -> no LDS/barriers
// in the loop. One cross-lane merge at the end.
// score(f,g) = sq[g] - 2*dot(A[f],A[g])  (monotone in distance per row)
// ---------------------------------------------------------------------------
__global__ __launch_bounds__(32)
void gram_topk(const __bf16* __restrict__ A, const float* __restrict__ sq,
               int* __restrict__ idxOut) {
    const int lane  = threadIdx.x;
    const int c     = blockIdx.y;
    const int fBase = blockIdx.x * 16;
    const int half  = lane >> 4;
    const int l15   = lane & 15;

    const __bf16* Ab  = A      + (size_t)c * F_FEAT * D_PAD;
    const float*  sqb = sq     + (size_t)c * F_FEAT;
    int*          idb = idxOut + (size_t)c * F_FEAT * 4;

    // Persistent A operand: full 16 x 320 tile (80 VGPRs)
    const __bf16* aRow = Ab + (size_t)(fBase + l15) * D_PAD + half * 8;
    v16bf av[K_STEPS];
#pragma unroll
    for (int s = 0; s < K_STEPS; ++s) av[s] = load_a_step(aRow, s * 32);

    // Per-lane top-4 for each of the 8 rows this lane owns in the D layout
    float ts[8][4];
    int   ti[8][4];
#pragma unroll
    for (int r = 0; r < 8; ++r)
#pragma unroll
        for (int t = 0; t < 4; ++t) { ts[r][t] = 3.4e38f; ti[r][t] = 0; }

    const __bf16* bBase = Ab + (size_t)l15 * D_PAD + half * 8;

    v16bf bv0[K_STEPS], bv1[K_STEPS];
    float sg0, sg1;

    // prologue: tile 0
    load_b_tile(bBase, bv0);
    sg0 = sqb[l15];

    for (int t = 0; t < N_TILES; t += 2) {
        // prefetch tile t+1 (always valid: N_TILES even)
        load_b_tile(bBase + (size_t)(t + 1) * 16 * D_PAD, bv1);
        sg1 = sqb[(t + 1) * 16 + l15];

        {   // compute + merge tile t (waits on loads issued one tile ago)
            v8f acc = {0.f, 0.f, 0.f, 0.f, 0.f, 0.f, 0.f, 0.f};
#pragma unroll
            for (int s = 0; s < K_STEPS; ++s)
                acc = __builtin_amdgcn_wmma_f32_16x16x32_bf16(
                    false, av[s], false, bv0[s], (short)0, acc, false, false);
            const int g = t * 16 + l15;
#pragma unroll
            for (int r = 0; r < 8; ++r)
                top4_insert(ts[r], ti[r], fmaf(-2.0f, acc[r], sg0), g);
        }

        // prefetch tile t+2 (clamped redundant prefetch on the last iteration)
        const int t2 = (t + 2 < N_TILES) ? (t + 2) : (N_TILES - 1);
        load_b_tile(bBase + (size_t)t2 * 16 * D_PAD, bv0);
        sg0 = sqb[t2 * 16 + l15];

        {   // compute + merge tile t+1
            v8f acc = {0.f, 0.f, 0.f, 0.f, 0.f, 0.f, 0.f, 0.f};
#pragma unroll
            for (int s = 0; s < K_STEPS; ++s)
                acc = __builtin_amdgcn_wmma_f32_16x16x32_bf16(
                    false, av[s], false, bv1[s], (short)0, acc, false, false);
            const int g = (t + 1) * 16 + l15;
#pragma unroll
            for (int r = 0; r < 8; ++r)
                top4_insert(ts[r], ti[r], fmaf(-2.0f, acc[r], sg1), g);
        }
    }

    // Cross-lane merge: row M is split over 16 lanes (one residue class each)
    __shared__ float smS[16][16][4];
    __shared__ int   smI[16][16][4];
#pragma unroll
    for (int r = 0; r < 8; ++r) {
        const int row = r + 8 * half;
#pragma unroll
        for (int t = 0; t < 4; ++t) {
            smS[row][l15][t] = ts[r][t];
            smI[row][l15][t] = ti[r][t];
        }
    }
    __syncthreads();

    if (lane < 16) {
        const int m = lane;
        float fs[4] = {3.4e38f, 3.4e38f, 3.4e38f, 3.4e38f};
        int   fi[4] = {0x7fffffff, 0x7fffffff, 0x7fffffff, 0x7fffffff};
        for (int src = 0; src < 16; ++src) {
#pragma unroll
            for (int t = 0; t < 4; ++t)
                top4_insert_lex(fs, fi, smS[m][src][t], smI[m][src][t]);
        }
        int* p = idb + (size_t)(fBase + m) * 4;
        p[0] = fi[0]; p[1] = fi[1]; p[2] = fi[2]; p[3] = fi[3];
    }
}

// ---------------------------------------------------------------------------
// K2: conv1 on X:  X1[b,o,f] = relu(b1[o] + sum_k x[b, idx1[f,k]] * w1[o,k])
// ---------------------------------------------------------------------------
__global__ void k2_conv1_x(const float* __restrict__ x, const int* __restrict__ idx1,
                           const float* __restrict__ w1, const float* __restrict__ b1,
                           float* __restrict__ X1) {
    __shared__ float sw[64];
    __shared__ float sb[16];
    if (threadIdx.x < 64) sw[threadIdx.x] = w1[threadIdx.x];
    if (threadIdx.x < 16) sb[threadIdx.x] = b1[threadIdx.x];
    __syncthreads();

    const int t = blockIdx.x * 256 + threadIdx.x;     // 128*2048 threads
    const int b = t >> 11;
    const int f = t & (F_FEAT - 1);

    float xv[4];
#pragma unroll
    for (int k = 0; k < 4; ++k) {
        int g = idx1[f * 4 + k];
        xv[k] = x[(size_t)b * F_FEAT + g];
    }
#pragma unroll
    for (int o = 0; o < NFILT; ++o) {
        float a = sb[o];
#pragma unroll
        for (int k = 0; k < 4; ++k) a += xv[k] * sw[o * 4 + k];
        X1[((size_t)(b * NFILT + o)) * F_FEAT + f] = fmaxf(a, 0.f);
    }
}

// ---------------------------------------------------------------------------
// K3: conv1 on crd -> bf16 planes Ac[c][f][d] (f-major, d padded) + sq_c[c][f]
// ---------------------------------------------------------------------------
__global__ __launch_bounds__(320)
void k3_conv1_crd(const float* __restrict__ coord, const int* __restrict__ idx1,
                  const float* __restrict__ w1, const float* __restrict__ b1,
                  __bf16* __restrict__ Ac, float* __restrict__ sqc) {
    __shared__ int   gidx[4];
    __shared__ float sw[64];
    __shared__ float sb[16];
    __shared__ float sqa[16];
    const int f = blockIdx.x;
    const int tid = threadIdx.x;
    if (tid < 4)  gidx[tid] = idx1[f * 4 + tid];
    if (tid < 64) sw[tid] = w1[tid];
    if (tid < 16) { sb[tid] = b1[tid]; sqa[tid] = 0.f; }
    __syncthreads();

    const int d = tid;                      // 0..319
    if (d < D_COORD) {
        float cv[4];
#pragma unroll
        for (int k = 0; k < 4; ++k) cv[k] = coord[(size_t)d * F_FEAT + gidx[k]];
#pragma unroll
        for (int c = 0; c < NFILT; ++c) {
            float v = sb[c];
#pragma unroll
            for (int k = 0; k < 4; ++k) v += cv[k] * sw[c * 4 + k];
            v = fmaxf(v, 0.f);
            unsigned short bb = f2bfbits(v);
            Ac[((size_t)c * F_FEAT + f) * D_PAD + d] = bfFromBits(bb);
            float q = bfBitsToF(bb);
            atomicAdd(&sqa[c], q * q);
        }
    } else {                                // pad d in [306,320)
#pragma unroll
        for (int c = 0; c < NFILT; ++c)
            Ac[((size_t)c * F_FEAT + f) * D_PAD + d] = bfFromBits(0);
    }
    __syncthreads();
    if (tid < 16) sqc[(size_t)tid * F_FEAT + f] = sqa[tid];
}

// ---------------------------------------------------------------------------
// K5: conv2 + fused 2-wide max-pool -> P bf16 (BATCH x DENSE_IN),
//     flat index o*1024 + fp (matches reshape/flatten in the reference)
// ---------------------------------------------------------------------------
__global__ void k5_conv2_pool(const float* __restrict__ X1, const int* __restrict__ idx2,
                              const float* __restrict__ w2, const float* __restrict__ b2,
                              __bf16* __restrict__ P) {
    __shared__ float sw[1024];
    __shared__ float sb[16];
    for (int i = threadIdx.x; i < 1024; i += 256) sw[i] = w2[i];
    if (threadIdx.x < 16) sb[threadIdx.x] = b2[threadIdx.x];
    __syncthreads();

    const int t  = blockIdx.x * 256 + threadIdx.x;   // 128*1024 threads
    const int b  = t >> 10;
    const int fp = t & 1023;

    float best[16];
#pragma unroll
    for (int half = 0; half < 2; ++half) {
        const int f = 2 * fp + half;
        float acc[16];
#pragma unroll
        for (int o = 0; o < 16; ++o) acc[o] = sb[o];
        for (int c = 0; c < 16; ++c) {
#pragma unroll
            for (int k = 0; k < 4; ++k) {
                int   g  = idx2[((size_t)(c * F_FEAT + f)) * 4 + k];
                float xv = X1[((size_t)(b * NFILT + c)) * F_FEAT + g];
#pragma unroll
                for (int o = 0; o < 16; ++o)
                    acc[o] += xv * sw[(o * 16 + c) * 4 + k];
            }
        }
#pragma unroll
        for (int o = 0; o < 16; ++o) {
            float r = fmaxf(acc[o], 0.f);
            best[o] = half ? fmaxf(best[o], r) : r;
        }
    }
#pragma unroll
    for (int o = 0; o < 16; ++o)
        P[(size_t)b * DENSE_IN + o * 1024 + fp] = f2bf(best[o]);
}

// ---------------------------------------------------------------------------
// K6a: dw1 f32 -> bf16
// ---------------------------------------------------------------------------
__global__ void k6a_cast(const float* __restrict__ src, __bf16* __restrict__ dst) {
    const int t = blockIdx.x * 256 + threadIdx.x;    // 64*16384 = 1048576
    dst[t] = f2bf(src[t]);
}

// ---------------------------------------------------------------------------
// K6: dense GEMM  H[b,j] = sum_k P[b,k] * W1[j,k]  via bf16 WMMA, K = 16384.
// 8 K-steps staged per chunk -> one 32-load clause, 8 back-to-back WMMAs.
// ---------------------------------------------------------------------------
__global__ __launch_bounds__(32)
void k6_dense_wmma(const __bf16* __restrict__ P, const __bf16* __restrict__ W,
                   float* __restrict__ H) {
    const int lane  = threadIdx.x;
    const int half  = lane >> 4;
    const int l15   = lane & 15;
    const int mBase = blockIdx.x * 16;    // batch tile (8)
    const int nBase = blockIdx.y * 16;    // hidden tile (4)

    const __bf16* aRow = P + (size_t)(mBase + l15) * DENSE_IN + half * 8;
    const __bf16* bRow = W + (size_t)(nBase + l15) * DENSE_IN + half * 8;

    v8f acc = {0.f, 0.f, 0.f, 0.f, 0.f, 0.f, 0.f, 0.f};
    for (int s0 = 0; s0 < DENSE_IN / 32; s0 += 8) {
        v16bf av[8], bv[8];
#pragma unroll
        for (int s = 0; s < 8; ++s) {
            const int ko = (s0 + s) * 32;
            av[s] = load_a_step(aRow, ko);
            bv[s] = load_b_step(bRow, ko);
        }
#pragma unroll
        for (int s = 0; s < 8; ++s)
            acc = __builtin_amdgcn_wmma_f32_16x16x32_bf16(
                false, av[s], false, bv[s], (short)0, acc, false, false);
    }
#pragma unroll
    for (int r = 0; r < 8; ++r) {
        const int M = r + half * 8;
        H[(size_t)(mBase + M) * HIDDEN + nBase + l15] = acc[r];
    }
}

// ---------------------------------------------------------------------------
// K7: out[b,n] = db2[n] + sum_j selu(H[b,j] + db1[j]) * dw2[n,j]
// ---------------------------------------------------------------------------
__global__ void k7_head(const float* __restrict__ H, const float* __restrict__ db1,
                        const float* __restrict__ dw2, const float* __restrict__ db2,
                        float* __restrict__ out) {
    const int t = blockIdx.x * 256 + threadIdx.x;
    if (t >= BATCH * NCLS) return;
    const int b = t / NCLS;
    const int n = t % NCLS;
    const float SC = 1.0507009873554804934193349852946f;
    const float AL = 1.6732632423543772848170429916717f;
    float acc = db2[n];
#pragma unroll 8
    for (int j = 0; j < HIDDEN; ++j) {
        float hv = H[(size_t)b * HIDDEN + j] + db1[j];
        hv = (hv > 0.f) ? SC * hv : SC * AL * (__expf(hv) - 1.f);
        acc += hv * dw2[n * HIDDEN + j];
    }
    out[(size_t)b * NCLS + n] = acc;
}

// ---------------------------------------------------------------------------
// Launcher: workspace layout (all 256B aligned), ~44 MB total.
// ---------------------------------------------------------------------------
extern "C" void kernel_launch(void* const* d_in, const int* in_sizes, int n_in,
                              void* d_out, int out_size, void* d_ws, size_t ws_size,
                              hipStream_t stream) {
    const float* x     = (const float*)d_in[0];
    const float* coord = (const float*)d_in[1];
    const float* w1    = (const float*)d_in[2];
    const float* b1    = (const float*)d_in[3];
    const float* w2    = (const float*)d_in[4];
    const float* b2    = (const float*)d_in[5];
    const float* dw1   = (const float*)d_in[6];
    const float* db1   = (const float*)d_in[7];
    const float* dw2   = (const float*)d_in[8];
    const float* db2   = (const float*)d_in[9];
    float* out = (float*)d_out;

    char* ws = (char*)d_ws;
    size_t off = 0;
    auto take = [&](size_t bytes) {
        void* p = ws + off;
        off = (off + bytes + 255) & ~(size_t)255;
        return p;
    };
    __bf16* A0   = (__bf16*)take((size_t)F_FEAT * D_PAD * 2);            // 1.25 MB
    float*  sq0  = (float*) take((size_t)F_FEAT * 4);
    int*    idx1 = (int*)   take((size_t)F_FEAT * 4 * 4);
    float*  X1   = (float*) take((size_t)BATCH * NFILT * F_FEAT * 4);    // 16 MB
    __bf16* Ac   = (__bf16*)take((size_t)NFILT * F_FEAT * D_PAD * 2);    // 20 MB
    float*  sqc  = (float*) take((size_t)NFILT * F_FEAT * 4);
    int*    idx2 = (int*)   take((size_t)NFILT * F_FEAT * 4 * 4);
    __bf16* P    = (__bf16*)take((size_t)BATCH * DENSE_IN * 2);          // 4 MB
    __bf16* W1b  = (__bf16*)take((size_t)HIDDEN * DENSE_IN * 2);         // 2 MB
    float*  H    = (float*) take((size_t)BATCH * HIDDEN * 4);

    // Stage 1: pack coord, Gram+top4 (c=1)
    k0_pack_coord<<<F_FEAT, 64, 0, stream>>>(coord, A0, sq0);
    gram_topk<<<dim3(F_FEAT / 16, 1), 32, 0, stream>>>(A0, sq0, idx1);

    // Stage 2: conv1 on X and crd
    k2_conv1_x<<<(BATCH * F_FEAT) / 256, 256, 0, stream>>>(x, idx1, w1, b1, X1);
    k3_conv1_crd<<<F_FEAT, 320, 0, stream>>>(coord, idx1, w1, b1, Ac, sqc);

    // Stage 3: the big one — 16 Gram matrices + fused top4 (~41 GFLOP)
    gram_topk<<<dim3(F_FEAT / 16, NFILT), 32, 0, stream>>>(Ac, sqc, idx2);

    // Stage 4: conv2 + maxpool -> bf16 P
    k5_conv2_pool<<<(BATCH * 1024) / 256, 256, 0, stream>>>(X1, idx2, w2, b2, P);

    // Stage 5: dense (WMMA) + head
    k6a_cast<<<(HIDDEN * DENSE_IN) / 256, 256, 0, stream>>>(dw1, W1b);
    k6_dense_wmma<<<dim3(BATCH / 16, HIDDEN / 16), 32, 0, stream>>>(P, W1b, H);
    k7_head<<<(BATCH * NCLS + 255) / 256, 256, 0, stream>>>(H, db1, dw2, db2, out);
}
